// VanillaGRU_71700184039683
// MI455X (gfx1250) — matrix operand order repaired and checked by reading, not compiled, not run on hardware
//
#include <hip/hip_runtime.h>

typedef __attribute__((ext_vector_type(16))) __bf16 v16bf;
typedef __attribute__((ext_vector_type(8)))  float  v8f;

#define GB   128   // batch
#define GT   2048  // timesteps
#define GIN  64    // input features (layer 0)
#define GH   256   // hidden
#define G3H  768   // 3*H (r,z,n gate rows)
#define GOUT 64    // output features

#define WMMA_BF16(a, b, c) \
  __builtin_amdgcn_wmma_f32_16x16x32_bf16(false, (a), false, (b), (short)0, (c), false, false)

// ---------- scalar helpers ----------
__device__ __forceinline__ unsigned short f2bf(float f) {
  union { float f; unsigned u; } v; v.f = f;
  unsigned r = v.u + 0x7FFFu + ((v.u >> 16) & 1u);   // round-to-nearest-even
  return (unsigned short)(r >> 16);
}
__device__ __forceinline__ float bf2f(unsigned short h) {
  union { unsigned u; float f; } v; v.u = ((unsigned)h) << 16;
  return v.f;
}
__device__ __forceinline__ float sigmf(float x) {
  return 1.0f / (1.0f + __expf(-x));
}
__device__ __forceinline__ float tanh_fast(float x) {
  float e = __expf(-2.0f * fabsf(x));
  float r = (1.0f - e) / (1.0f + e);
  return copysignf(r, x);
}

// ---------- WMMA fragment loaders ----------
union FragU { uint4 u[2]; v16bf v; };

// A (16xK bf16) from LDS. CDNA5 16-bit A layout:
// lane m=l%16, half=l/16; VGPR0-3: K = k0+8*half+{0..7}; VGPR4-7: K = k0+16+8*half+{0..7}
__device__ __forceinline__ v16bf load_a_lds(const unsigned short* buf, int lda,
                                            int k0, int m, int half) {
  const unsigned short* p = buf + m * lda + k0 + 8 * half;
  FragU r;
  r.u[0] = *(const uint4*)(p);        // 8 bf16: K = k0+8h .. +7
  r.u[1] = *(const uint4*)(p + 16);   // 8 bf16: K = k0+16+8h .. +7
  return r.v;
}

// B (32x16 bf16) from global, weights pre-transposed to [K][N] bf16.
// CDNA5 16-bit B layout: lane column N=n0+l%16; K = k0+8*half+{0..7}, then k0+16+8*half+{0..7}.
__device__ __forceinline__ v16bf load_b_glb(const unsigned short* Wt, int ldn,
                                            int k0, int n0, int lane) {
  const unsigned short* p = Wt + (size_t)(k0 + 8 * (lane >> 4)) * ldn + n0 + (lane & 15);
  union { unsigned short s[16]; v16bf v; } r;
#pragma unroll
  for (int s = 0; s < 8; ++s) {
    r.s[s]     = p[(size_t)s * ldn];
    r.s[s + 8] = p[(size_t)(s + 16) * ldn];
  }
  return r.v;
}

// One gate-projection GEMM phase: acc{R,Z,N} += A[16xK] * Wt[Kx768] tiles for
// this wave's 16 hidden columns in each of the r/z/n gate blocks.
__device__ __forceinline__ void gate_gemm(const unsigned short* Abuf, int lda, int K,
                                          const unsigned short* Wt,
                                          int wv, int lane, int m, int half,
                                          v8f& accR, v8f& accZ, v8f& accN) {
  for (int k0 = 0; k0 < K; k0 += 32) {
    v16bf a  = load_a_lds(Abuf, lda, k0, m, half);
    v16bf b0 = load_b_glb(Wt, G3H, k0, 0 * GH + 16 * wv, lane);
    accR = WMMA_BF16(a, b0, accR);
    v16bf b1 = load_b_glb(Wt, G3H, k0, 1 * GH + 16 * wv, lane);
    accZ = WMMA_BF16(a, b1, accZ);
    v16bf b2 = load_b_glb(Wt, G3H, k0, 2 * GH + 16 * wv, lane);
    accN = WMMA_BF16(a, b2, accN);
  }
}

// ---------- weight convert + transpose: fp32 [N][K] -> bf16 [K][N] ----------
__global__ void wconv_kernel(const float* __restrict__ W, unsigned short* __restrict__ Wt,
                             int N, int K) {
  int i = blockIdx.x * blockDim.x + threadIdx.x;
  if (i >= N * K) return;
  int k = i / N;
  int n = i - k * N;
  Wt[i] = f2bf(W[(size_t)n * K + k]);
}

// ---------- persistent 2-layer GRU + output linear ----------
// grid = B/16 blocks (independent batch tiles), 512 threads = 16 waves.
// Wave w owns hidden columns [16w, 16w+16): one 16x16 N-tile per gate block.
// NOTE: weight pointers are intentionally NOT __restrict__ and a compiler
// memory barrier sits at the top of the t-loop, so the B-fragment loads stay
// inside the loop (streamed from L2 each step) instead of being hoisted and
// spilled to scratch.
__global__ __launch_bounds__(512) void gru2_persistent(
    const float* __restrict__ x,
    const unsigned short* Wih0t,   // [64 ][768] bf16
    const unsigned short* Whh0t,   // [256][768]
    const unsigned short* Wih1t,   // [256][768]
    const unsigned short* Whh1t,   // [256][768]
    const unsigned short* Woutt,   // [256][64 ]
    const float* __restrict__ bih0, const float* __restrict__ bhh0,
    const float* __restrict__ bih1, const float* __restrict__ bhh1,
    const float* __restrict__ bout,
    float* out)
{
  __shared__ unsigned short xbuf [16 * GIN];  // x_t tile, bf16
  __shared__ unsigned short h0buf[16 * GH];   // layer0 hidden state, bf16
  __shared__ unsigned short h1buf[16 * GH];   // layer1 hidden state, bf16

  const int tid  = threadIdx.x;
  const int lane = tid & 31;
  const int wv   = tid >> 5;          // wave 0..15
  const int m    = lane & 15;
  const int half = lane >> 4;
  const int b0   = blockIdx.x * 16;   // batch tile base
  const int colw = 16 * wv + m;       // this lane's hidden column

  for (int i = tid; i < 16 * GH; i += 512) { h0buf[i] = 0; h1buf[i] = 0; }

  // per-lane gate biases; r/z biases pre-summed (x-side + h-side share acc)
  float br0 = bih0[0 * GH + colw] + bhh0[0 * GH + colw];
  float bz0 = bih0[1 * GH + colw] + bhh0[1 * GH + colw];
  float bxn0 = bih0[2 * GH + colw], bhn0 = bhh0[2 * GH + colw];
  float br1 = bih1[0 * GH + colw] + bhh1[0 * GH + colw];
  float bz1 = bih1[1 * GH + colw] + bhh1[1 * GH + colw];
  float bxn1 = bih1[2 * GH + colw], bhn1 = bhh1[2 * GH + colw];
  const float bo = (wv < 4) ? bout[16 * wv + m] : 0.0f;

  __syncthreads();

  for (int t = 0; t < GT; ++t) {
    // prevent cross-iteration CSE / hoisting of weight loads
    asm volatile("" ::: "memory");

    // -------- stage x_t tile -> LDS bf16 (512 threads, 2 elems each) --------
    {
      int row = tid >> 5;             // 0..15
      int c0  = (tid & 31) * 2;       // 0..62
      const float2 v = *(const float2*)(x + ((size_t)(b0 + row) * GT + t) * GIN + c0);
      xbuf[row * GIN + c0 + 0] = f2bf(v.x);
      xbuf[row * GIN + c0 + 1] = f2bf(v.y);
    }
    __syncthreads();

    // ================= layer 0 =================
    {
      v8f accR = {}, accZ = {}, accXN = {}, accHN = {};
      gate_gemm(xbuf,  GIN, GIN, Wih0t, wv, lane, m, half, accR, accZ, accXN);
      gate_gemm(h0buf, GH,  GH,  Whh0t, wv, lane, m, half, accR, accZ, accHN);
      float hold[8];
#pragma unroll
      for (int k = 0; k < 8; ++k) hold[k] = bf2f(h0buf[(k + 8 * half) * GH + colw]);
      __syncthreads();                                  // all reads of old h done
#pragma unroll
      for (int k = 0; k < 8; ++k) {
        float r = sigmf(accR[k] + br0);
        float z = sigmf(accZ[k] + bz0);
        float n = tanh_fast(accXN[k] + bxn0 + r * (accHN[k] + bhn0));
        float hn = (1.0f - z) * n + z * hold[k];
        h0buf[(k + 8 * half) * GH + colw] = f2bf(hn);
      }
      __syncthreads();
    }

    // ================= layer 1 (input = h0buf) =================
    {
      v8f accR = {}, accZ = {}, accXN = {}, accHN = {};
      gate_gemm(h0buf, GH, GH, Wih1t, wv, lane, m, half, accR, accZ, accXN);
      gate_gemm(h1buf, GH, GH, Whh1t, wv, lane, m, half, accR, accZ, accHN);
      float hold[8];
#pragma unroll
      for (int k = 0; k < 8; ++k) hold[k] = bf2f(h1buf[(k + 8 * half) * GH + colw]);
      __syncthreads();
#pragma unroll
      for (int k = 0; k < 8; ++k) {
        float r = sigmf(accR[k] + br1);
        float z = sigmf(accZ[k] + bz1);
        float n = tanh_fast(accXN[k] + bxn1 + r * (accHN[k] + bhn1));
        float hn = (1.0f - z) * n + z * hold[k];
        h1buf[(k + 8 * half) * GH + colw] = f2bf(hn);
      }
      __syncthreads();
    }

    // ================= output linear: out_t = h1 @ Wout^T + b =================
    if (wv < 4) {                       // 4 N-tiles cover OUT=64
      v8f acc = {};
      for (int k0 = 0; k0 < GH; k0 += 32) {
        v16bf a = load_a_lds(h1buf, GH, k0, m, half);
        v16bf b = load_b_glb(Woutt, GOUT, k0, 16 * wv, lane);
        acc = WMMA_BF16(a, b, acc);
      }
      const int o = 16 * wv + m;
#pragma unroll
      for (int k = 0; k < 8; ++k) {
        int row = k + 8 * half;
        out[((size_t)(b0 + row) * GT + t) * GOUT + o] = acc[k] + bo;
      }
    }
    __syncthreads();
  }
}

// ---------- host-side launcher ----------
extern "C" void kernel_launch(void* const* d_in, const int* in_sizes, int n_in,
                              void* d_out, int out_size, void* d_ws, size_t ws_size,
                              hipStream_t stream) {
  const float* x    = (const float*)d_in[0];
  const float* Wih0 = (const float*)d_in[1];
  const float* Whh0 = (const float*)d_in[2];
  const float* bih0 = (const float*)d_in[3];
  const float* bhh0 = (const float*)d_in[4];
  const float* Wih1 = (const float*)d_in[5];
  const float* Whh1 = (const float*)d_in[6];
  const float* bih1 = (const float*)d_in[7];
  const float* bhh1 = (const float*)d_in[8];
  const float* Wout = (const float*)d_in[9];
  const float* bout = (const float*)d_in[10];
  float* out = (float*)d_out;

  // workspace: bf16 transposed weights (~1.3 MB total)
  unsigned short* ws    = (unsigned short*)d_ws;
  unsigned short* Wih0t = ws;                       // 64  x 768
  unsigned short* Whh0t = Wih0t + 64  * 768;        // 256 x 768
  unsigned short* Wih1t = Whh0t + 256 * 768;        // 256 x 768
  unsigned short* Whh1t = Wih1t + 256 * 768;        // 256 x 768
  unsigned short* Woutt = Whh1t + 256 * 768;        // 256 x 64

  auto conv = [&](const float* W, unsigned short* Wt, int N, int K) {
    int n = N * K;
    wconv_kernel<<<(n + 255) / 256, 256, 0, stream>>>(W, Wt, N, K);
  };
  conv(Wih0, Wih0t, G3H, GIN);
  conv(Whh0, Whh0t, G3H, GH);
  conv(Wih1, Wih1t, G3H, GH);
  conv(Whh1, Whh1t, G3H, GH);
  conv(Wout, Woutt, GOUT, GH);

  gru2_persistent<<<GB / 16, 512, 0, stream>>>(
      x, Wih0t, Whh0t, Wih1t, Whh1t, Woutt,
      bih0, bhh0, bih1, bhh1, bout, out);
}
